// MultiHeadAttention_28114855920391
// MI455X (gfx1250) — matrix operand order repaired and checked by reading, not compiled, not run on hardware
//
#include <hip/hip_runtime.h>

// ---------------------------------------------------------------------------
// MultiHeadAttention for MI455X (gfx1250): bf16 WMMA GEMMs + flash-style
// fused softmax-over-query attention.  B=4, S=1024, D=256, H=8.
// ---------------------------------------------------------------------------

typedef __bf16 bf16;
typedef bf16 v16bf __attribute__((ext_vector_type(16)));
typedef float v8f  __attribute__((ext_vector_type(8)));

#define BB   4
#define SS   1024
#define DD   256
#define HH   8
#define NDH  2048   // D*H
#define ROWS 4096   // B*S

// LDS row paddings (elements) chosen for conflict-free ds_load_b128:
//  K tile rows: 264 elems -> 132 dwords stride; 132 % 64 = 4 -> 16 lanes
//  land on distinct bank quads.  P rows: 40 elems -> 20 dwords stride,
//  lm*20 % 64 all distinct multiples of 4.
#define KPAD 264
#define PPAD 40

union Frag { uint4 u[2]; v16bf v; };

// A-matrix fragment (16x32 bf16, M x K): lane holds row m = lane%16,
// half hh = lane/16; dwords 0..3 hold k = hh*8 .. hh*8+7,
// dwords 4..7 hold k = 16+hh*8 .. +7  (ISA 7.12.2, 16-bit A 16x32).
__device__ __forceinline__ v16bf load_frag_a(const bf16* row, int hh) {
  Frag f;
  f.u[0] = *(const uint4*)(row + hh * 8);
  f.u[1] = *(const uint4*)(row + 16 + hh * 8);
  return f.v;
}

// B-matrix fragment (32x16 bf16, K x N) from a row-major *transposed* source
// (row n, contiguous k): lane holds column n = lane%16, dwords 0..3 hold
// k = hh*8 .. hh*8+7, dwords 4..7 hold k = 16+hh*8 .. +7 (matches A).
__device__ __forceinline__ v16bf load_frag_b(const bf16* row, int hh) {
  Frag f;
  f.u[0] = *(const uint4*)(row + hh * 8);
  f.u[1] = *(const uint4*)(row + 16 + hh * 8);
  return f.v;
}

__device__ __forceinline__ v8f wmma_bf16(v16bf a, v16bf b, v8f c) {
  return __builtin_amdgcn_wmma_f32_16x16x32_bf16(
      /*neg_a=*/false, a, /*neg_b=*/false, b,
      /*c_mod=*/(short)0, c, /*reuse_a=*/false, /*reuse_b=*/false);
}

// ---------------------------------------------------------------------------
__global__ __launch_bounds__(256) void cvt_f32_bf16(
    const float* __restrict__ src, bf16* __restrict__ dst, int n) {
  int i = blockIdx.x * blockDim.x + threadIdx.x;
  if (i < n) dst[i] = (bf16)src[i];
}

// dst[n*K + k] = src[k*N + n]   (transpose + convert, src is K x N f32)
__global__ __launch_bounds__(256) void transpose_f32_bf16(
    const float* __restrict__ src, bf16* __restrict__ dst, int K, int N) {
  int i = blockIdx.x * blockDim.x + threadIdx.x;
  if (i < K * N) {
    int n = i / K;
    int k = i - n * K;
    dst[i] = (bf16)src[(size_t)k * N + n];
  }
}

// ---------------------------------------------------------------------------
// Projection GEMM: out(ROWS x NDH) = A(ROWS x DD) @ W(DD x NDH) + bias.
// Bt is W transposed (NDH x DD, row-major).  Wave tile: 16(M) x 64(N).
// MODE 0: bf16 row-major [ROWS][NDH];  MODE 1: scatter into [B][H][D][S].
template <int MODE>
__global__ __launch_bounds__(128) void gemm_proj16x64(
    const bf16* __restrict__ A, const bf16* __restrict__ Bt,
    const float* __restrict__ bias, bf16* __restrict__ outb) {
  const int K = DD, N = NDH;
  int lane = threadIdx.x & 31, wave = threadIdx.x >> 5;
  int lm = lane & 15, hh = lane >> 4;
  int m0 = blockIdx.x * 16;
  int n0 = blockIdx.y * 256 + wave * 64;
  const bf16* arow = A + (size_t)(m0 + lm) * K;

  v8f acc[4];
#pragma unroll
  for (int t = 0; t < 4; t++)
#pragma unroll
    for (int i = 0; i < 8; i++) acc[t][i] = 0.f;

#pragma unroll
  for (int k0 = 0; k0 < K; k0 += 32) {
    v16bf af = load_frag_a(arow + k0, hh);
#pragma unroll
    for (int t = 0; t < 4; t++) {
      v16bf bf = load_frag_b(Bt + (size_t)(n0 + t * 16 + lm) * K + k0, hh);
      acc[t] = wmma_bf16(af, bf, acc[t]);
    }
  }

#pragma unroll
  for (int t = 0; t < 4; t++)
#pragma unroll
    for (int v = 0; v < 8; v++) {
      int m = m0 + v + 8 * hh;        // C/D layout: vgpr v + half -> row
      int n = n0 + t * 16 + lm;       // lane%16 -> column
      float val = acc[t][v] + bias[n];
      if (MODE == 0) {
        outb[(size_t)m * N + n] = (bf16)val;
      } else {
        int b    = m >> 10, qpos = m & (SS - 1);
        int head = n >> 8,  d    = n & (DD - 1);
        outb[((size_t)(b * HH + head) * DD + d) * SS + qpos] = (bf16)val;
      }
    }
}

// ---------------------------------------------------------------------------
// Flash attention with softmax over the QUERY axis:
//   out[k,d] = sum_q softmax_q(scale * K[k]·Q[q]) * V[q,d]
// One wave owns 16 key rows x full d=256 (16 accum tiles), streams q by 32.
// K-tile lives in LDS (removes the VGPR pressure that caused scratch spills).
__global__ __launch_bounds__(128, 1) void flash_attn(
    const bf16* __restrict__ Kp, const bf16* __restrict__ Qp,
    const bf16* __restrict__ Vt, bf16* __restrict__ Oh) {
  __shared__ __attribute__((aligned(16))) bf16 klds[4][16][KPAD];
  __shared__ __attribute__((aligned(16))) bf16 plds[4][16][PPAD];
  int lane = threadIdx.x & 31, wave = threadIdx.x >> 5;
  int lm = lane & 15, hh = lane >> 4;
  int tile = blockIdx.x * 4 + wave;     // 2048 tiles = B*H*(S/16)
  int b   = tile >> 9;
  int rem = tile & 511;
  int h_  = rem >> 6;                   // head
  int kt  = rem & 63;                   // key tile

  // ---- stage this wave's 16x256 K tile into LDS (coalesced 512B rows) ----
  {
    const bf16* kbase = Kp + (size_t)(b * SS + kt * 16) * NDH + h_ * DD;
#pragma unroll
    for (int r = 0; r < 16; r++) {
      uint4 d = *(const uint4*)(kbase + (size_t)r * NDH + lane * 8);
      *(uint4*)&klds[wave][r][lane * 8] = d;
    }
  }
  asm volatile("s_wait_dscnt 0" ::: "memory");  // same-wave LDS RAW

  const bf16* vbase = Vt + ((size_t)(b * HH + h_) * DD + lm) * SS;

  v8f acc[16];
#pragma unroll
  for (int t = 0; t < 16; t++)
#pragma unroll
    for (int i = 0; i < 8; i++) acc[t][i] = 0.f;

  float mrow[8], lrow[8];
#pragma unroll
  for (int v = 0; v < 8; v++) { mrow[v] = -__builtin_inff(); lrow[v] = 0.f; }
  const float scale = 0.0625f;  // 1/sqrt(256)

  for (int q0 = 0; q0 < SS; q0 += 32) {
    // ---- S tiles: S[m=key, n=query] over the d=256 contraction ----
    v8f s0, s1;
#pragma unroll
    for (int i = 0; i < 8; i++) { s0[i] = 0.f; s1[i] = 0.f; }
    const bf16* qrow0 = Qp + (size_t)(b * SS + q0 + lm) * NDH + h_ * DD;
#pragma unroll
    for (int kk = 0; kk < DD; kk += 32) {
      Frag af;  // A-fragment of K from LDS
      af.u[0] = *(const uint4*)&klds[wave][lm][kk + hh * 8];
      af.u[1] = *(const uint4*)&klds[wave][lm][kk + 16 + hh * 8];
      v16bf b0 = load_frag_b(qrow0 + kk, hh);
      v16bf b1 = load_frag_b(qrow0 + (size_t)16 * NDH + kk, hh);
      s0 = wmma_bf16(af.v, b0, s0);
      s1 = wmma_bf16(af.v, b1, s1);
    }

    // ---- online softmax across q (lane groups of 16 within each half) ----
    float p0[8], p1[8];
#pragma unroll
    for (int v = 0; v < 8; v++) {
      float x0 = s0[v] * scale, x1 = s1[v] * scale;
      float mx = fmaxf(x0, x1);
      mx = fmaxf(mx, __shfl_xor(mx, 1, 32));
      mx = fmaxf(mx, __shfl_xor(mx, 2, 32));
      mx = fmaxf(mx, __shfl_xor(mx, 4, 32));
      mx = fmaxf(mx, __shfl_xor(mx, 8, 32));
      float mnew = fmaxf(mrow[v], mx);
      float corr = __expf(mrow[v] - mnew);
      float e0 = __expf(x0 - mnew);
      float e1 = __expf(x1 - mnew);
      float sum = e0 + e1;
      sum += __shfl_xor(sum, 1, 32);
      sum += __shfl_xor(sum, 2, 32);
      sum += __shfl_xor(sum, 4, 32);
      sum += __shfl_xor(sum, 8, 32);
      lrow[v] = lrow[v] * corr + sum;
      mrow[v] = mnew;
      p0[v] = e0; p1[v] = e1;
#pragma unroll
      for (int t = 0; t < 16; t++) acc[t][v] *= corr;  // rescale row m=v+8h
    }

    // ---- D-layout P -> A-fragment layout via per-wave LDS staging ----
#pragma unroll
    for (int v = 0; v < 8; v++) {
      int m = v + 8 * hh;
      plds[wave][m][lm]      = (bf16)p0[v];
      plds[wave][m][16 + lm] = (bf16)p1[v];
    }
    asm volatile("s_wait_dscnt 0" ::: "memory");  // same-wave LDS RAW
    Frag pf;
    pf.u[0] = *(const uint4*)&plds[wave][lm][hh * 8];
    pf.u[1] = *(const uint4*)&plds[wave][lm][16 + hh * 8];

    // ---- acc += P(16x32) @ V(32 x 256), 16 n-tiles of d ----
#pragma unroll
    for (int t = 0; t < 16; t++) {
      v16bf vf = load_frag_b(vbase + (size_t)t * 16 * SS + q0, hh);
      acc[t] = wmma_bf16(pf.v, vf, acc[t]);
    }
  }

  // ---- normalize and store bf16 into Oh[ROWS][NDH] ----
#pragma unroll
  for (int v = 0; v < 8; v++) {
    float inv = 1.0f / lrow[v];
    int m = v + 8 * hh;
    size_t row = (size_t)(b * SS + kt * 16 + m) * NDH + h_ * DD;
#pragma unroll
    for (int t = 0; t < 16; t++)
      Oh[row + t * 16 + lm] = (bf16)(acc[t][v] * inv);
  }
}

// ---------------------------------------------------------------------------
// Output GEMM: out(ROWS x DD) f32 = Oh(ROWS x NDH) @ Wo(NDH x DD) + bo
__global__ __launch_bounds__(128) void gemm_out16x64(
    const bf16* __restrict__ A, const bf16* __restrict__ Bt,  // WoT: DD x NDH
    const float* __restrict__ bias, float* __restrict__ out) {
  const int K = NDH, N = DD;
  int lane = threadIdx.x & 31, wave = threadIdx.x >> 5;
  int lm = lane & 15, hh = lane >> 4;
  int m0 = blockIdx.x * 16;
  int n0 = wave * 64;
  const bf16* arow = A + (size_t)(m0 + lm) * K;

  v8f acc[4];
#pragma unroll
  for (int t = 0; t < 4; t++)
#pragma unroll
    for (int i = 0; i < 8; i++) acc[t][i] = 0.f;

  for (int k0 = 0; k0 < K; k0 += 32) {
    v16bf af = load_frag_a(arow + k0, hh);
#pragma unroll
    for (int t = 0; t < 4; t++) {
      v16bf bf = load_frag_b(Bt + (size_t)(n0 + t * 16 + lm) * K + k0, hh);
      acc[t] = wmma_bf16(af, bf, acc[t]);
    }
  }

#pragma unroll
  for (int t = 0; t < 4; t++)
#pragma unroll
    for (int v = 0; v < 8; v++) {
      int m = m0 + v + 8 * hh;
      int n = n0 + t * 16 + lm;
      out[(size_t)m * N + n] = acc[t][v] + bias[n];
    }
}

// ---------------------------------------------------------------------------
extern "C" void kernel_launch(void* const* d_in, const int* in_sizes, int n_in,
                              void* d_out, int out_size, void* d_ws,
                              size_t ws_size, hipStream_t stream) {
  (void)in_sizes; (void)n_in; (void)out_size; (void)ws_size;
  const float* value_in = (const float*)d_in[0];
  const float* key_in   = (const float*)d_in[1];
  const float* query_in = (const float*)d_in[2];
  const float* Wv = (const float*)d_in[3];
  const float* bv = (const float*)d_in[4];
  const float* Wk = (const float*)d_in[5];
  const float* bk = (const float*)d_in[6];
  const float* Wq = (const float*)d_in[7];
  const float* bq = (const float*)d_in[8];
  const float* Wo = (const float*)d_in[9];
  const float* bo = (const float*)d_in[10];
  float* out = (float*)d_out;

  char* p = (char*)d_ws;
  auto alloc = [&](size_t elems) {
    bf16* r = (bf16*)p;
    p += ((elems * sizeof(bf16) + 255) & ~(size_t)255);
    return r;
  };
  bf16* Xv  = alloc((size_t)ROWS * DD);
  bf16* Xk  = alloc((size_t)ROWS * DD);
  bf16* Xq  = alloc((size_t)ROWS * DD);
  bf16* WvT = alloc((size_t)NDH * DD);
  bf16* WkT = alloc((size_t)NDH * DD);
  bf16* WqT = alloc((size_t)NDH * DD);
  bf16* WoT = alloc((size_t)DD * NDH);
  bf16* Qp  = alloc((size_t)ROWS * NDH);
  bf16* Kp  = alloc((size_t)ROWS * NDH);
  bf16* Vt  = alloc((size_t)ROWS * NDH);   // [B][H][D][S] transposed layout
  bf16* Oh  = alloc((size_t)ROWS * NDH);

  const int nIn = ROWS * DD;               // 1,048,576
  cvt_f32_bf16<<<(nIn + 255) / 256, 256, 0, stream>>>(value_in, Xv, nIn);
  cvt_f32_bf16<<<(nIn + 255) / 256, 256, 0, stream>>>(key_in,   Xk, nIn);
  cvt_f32_bf16<<<(nIn + 255) / 256, 256, 0, stream>>>(query_in, Xq, nIn);

  const int nW = DD * NDH;                 // 524,288
  transpose_f32_bf16<<<(nW + 255) / 256, 256, 0, stream>>>(Wv, WvT, DD, NDH);
  transpose_f32_bf16<<<(nW + 255) / 256, 256, 0, stream>>>(Wk, WkT, DD, NDH);
  transpose_f32_bf16<<<(nW + 255) / 256, 256, 0, stream>>>(Wq, WqT, DD, NDH);
  transpose_f32_bf16<<<(nW + 255) / 256, 256, 0, stream>>>(Wo, WoT, NDH, DD);

  dim3 gProj(ROWS / 16, NDH / 256);        // 256 x 8 blocks, 4 waves each
  gemm_proj16x64<0><<<gProj, 128, 0, stream>>>(Xq, WqT, bq, Qp);
  gemm_proj16x64<0><<<gProj, 128, 0, stream>>>(Xk, WkT, bk, Kp);
  gemm_proj16x64<1><<<gProj, 128, 0, stream>>>(Xv, WvT, bv, Vt);

  flash_attn<<<(BB * HH * (SS / 16)) / 4, 128, 0, stream>>>(Kp, Qp, Vt, Oh);

  gemm_out16x64<<<ROWS / 16, 128, 0, stream>>>(Oh, WoT, bo, out);
}
